// SolitonInteractionLayer_81853486727569
// MI455X (gfx1250) — compile-verified, matrix operand order checked by reading, not run on hardware
//
#include <hip/hip_runtime.h>

// ---------------------------------------------------------------------------
// SolitonInteractionLayer for MI455X (gfx1250, wave32)
//   Kernel 1: W f32 -> f16
//   Kernel 2: fused tanh + 3x RK4 PDE solvers per (b,d) column, LDS stencil
//   Kernel 3: WMMA f16 GEMM (evolved[M,K] x W[N,K]^T + bias), f32 out
// ---------------------------------------------------------------------------

typedef __attribute__((ext_vector_type(16))) _Float16 v16h;
typedef __attribute__((ext_vector_type(8)))  _Float16 v8h;
typedef __attribute__((ext_vector_type(8)))  float    v8f;

#define B_DIM 4
#define T_DIM 4096
#define D_DIM 1024
#define NTH   256
#define EPT   (T_DIM / NTH)   // 16 elements per thread along T
#define DT0   0.05f
#define DAMPC 0.5f

// ---------------------------- PDE stage RHS functors -----------------------
// LDS array s holds the current stage's u with 4 leading zeros so that
// causal shifts (zero padding) fall out automatically: u[t-j] == s[4+t-j].

struct SGRhs {  // C2*dxx(u) - sin(u) - DAMP*v      (C2 == 1)
    __device__ __forceinline__ float operator()(const float* s, int t, float vc) const {
        float uc = s[4 + t];
        return (uc - 2.0f * s[3 + t] + s[2 + t]) - __sinf(uc) - DAMPC * vc;
    }
};

struct HJRhs {  // dx((1+u+u^2)*dx(u)) - dxxxx(u) - DAMP*v
    __device__ __forceinline__ float operator()(const float* s, int t, float vc) const {
        float uc = s[4 + t], u1 = s[3 + t], u2 = s[2 + t], u3 = s[1 + t], u4 = s[0 + t];
        float g  = (1.0f + uc + uc * uc) * (uc - u1);
        float g1 = (1.0f + u1 + u1 * u1) * (u1 - u2);
        float d4 = uc - 4.0f * u1 + 6.0f * u2 - 4.0f * u3 + u4;
        return (g - g1) - d4 - DAMPC * vc;
    }
};

struct KdVRhs { // -6u*dx(u) - dxxx(u) - DAMP*u
    __device__ __forceinline__ float operator()(const float* s, int t) const {
        float uc = s[4 + t], u1 = s[3 + t], u2 = s[2 + t], u3 = s[1 + t];
        float dx1 = uc - u1;
        float d3  = uc - 3.0f * u1 + 3.0f * u2 - u3;
        return -6.0f * uc * dx1 - d3 - DAMPC * uc;
    }
};

// ------------------------------ RK4 drivers --------------------------------

template <typename RHS>
__device__ __forceinline__ void rk4_2nd(float* u, float dt, int nsteps,
                                        float* s_u, int tid, RHS rhs) {
    float v[EPT], us[EPT], vs[EPT], kv[EPT], au[EPT], av[EPT];
#pragma unroll
    for (int i = 0; i < EPT; ++i) v[i] = 0.0f;
    for (int step = 0; step < nsteps; ++step) {
#pragma unroll
        for (int i = 0; i < EPT; ++i) { us[i] = u[i]; vs[i] = v[i]; au[i] = 0.0f; av[i] = 0.0f; }
#pragma unroll
        for (int st = 0; st < 4; ++st) {
            __syncthreads();                    // previous stage reads done
#pragma unroll
            for (int i = 0; i < EPT; ++i) s_u[4 + tid + i * NTH] = us[i];
            __syncthreads();                    // stage field visible
#pragma unroll
            for (int i = 0; i < EPT; ++i) kv[i] = rhs(s_u, tid + i * NTH, vs[i]);
            const float wgt = (st == 0 || st == 3) ? 1.0f : 2.0f;
            const float c   = (st < 2) ? 0.5f * dt : dt;
#pragma unroll
            for (int i = 0; i < EPT; ++i) {
                au[i] += wgt * vs[i];           // k_u of this stage is vs
                av[i] += wgt * kv[i];
                float nus = u[i] + c * vs[i];   // build next stage (unused after st==3)
                vs[i] = v[i] + c * kv[i];
                us[i] = nus;
            }
        }
#pragma unroll
        for (int i = 0; i < EPT; ++i) {
            u[i] += (dt / 6.0f) * au[i];
            v[i] += (dt / 6.0f) * av[i];
        }
    }
}

template <typename RHS>
__device__ __forceinline__ void rk4_1st(float* u, float dt, int nsteps,
                                        float* s_u, int tid, RHS rhs) {
    float us[EPT], k[EPT], acc[EPT];
    for (int step = 0; step < nsteps; ++step) {
#pragma unroll
        for (int i = 0; i < EPT; ++i) { us[i] = u[i]; acc[i] = 0.0f; }
#pragma unroll
        for (int st = 0; st < 4; ++st) {
            __syncthreads();
#pragma unroll
            for (int i = 0; i < EPT; ++i) s_u[4 + tid + i * NTH] = us[i];
            __syncthreads();
#pragma unroll
            for (int i = 0; i < EPT; ++i) k[i] = rhs(s_u, tid + i * NTH);
            const float wgt = (st == 0 || st == 3) ? 1.0f : 2.0f;
            const float c   = (st < 2) ? 0.5f * dt : dt;
#pragma unroll
            for (int i = 0; i < EPT; ++i) {
                acc[i] += wgt * k[i];
                us[i] = u[i] + c * k[i];
            }
        }
#pragma unroll
        for (int i = 0; i < EPT; ++i) u[i] += (dt / 6.0f) * acc[i];
    }
}

// ------------------------- Kernel 2: fused PDE evolution --------------------
// One block per (b,d) column: full T=4096 sequence, zero halo recompute.
// Stage field lives in LDS (16.4 KB); all other RK4 state in VGPRs.

__global__ __launch_bounds__(NTH) void pde_kernel(const float* __restrict__ x,
                                                  const float* __restrict__ mix,
                                                  _Float16* __restrict__ evh) {
    __shared__ float s_u[T_DIM + 8];
    const int tid = threadIdx.x;
    const int bid = blockIdx.x;
    const int b = bid >> 10;            // / D_DIM
    const int d = bid & (D_DIM - 1);

    if (tid < 4) s_u[tid] = 0.0f;       // causal zero padding

    // softmax(solver_mix)
    float m0 = mix[0], m1 = mix[1], m2 = mix[2];
    float mx = fmaxf(m0, fmaxf(m1, m2));
    float e0 = __expf(m0 - mx), e1 = __expf(m1 - mx), e2 = __expf(m2 - mx);
    float inv = 1.0f / (e0 + e1 + e2);
    float w0 = e0 * inv, w1 = e1 * inv, w2 = e2 * inv;

    const float* xcol = x + (size_t)b * T_DIM * D_DIM + d;

    float u0[EPT], ua[EPT], acc[EPT];
#pragma unroll
    for (int i = 0; i < EPT; ++i)
        u0[i] = 2.0f * tanhf(xcol[(size_t)(tid + i * NTH) * D_DIM]);

    // sine-Gordon: dt = 0.05
#pragma unroll
    for (int i = 0; i < EPT; ++i) ua[i] = u0[i];
    rk4_2nd(ua, DT0, 3, s_u, tid, SGRhs{});
#pragma unroll
    for (int i = 0; i < EPT; ++i) acc[i] = w0 * ua[i];

    // KdV: dt = 0.025, first order
#pragma unroll
    for (int i = 0; i < EPT; ++i) ua[i] = u0[i];
    rk4_1st(ua, 0.5f * DT0, 3, s_u, tid, KdVRhs{});
#pragma unroll
    for (int i = 0; i < EPT; ++i) acc[i] += w1 * ua[i];

    // Heimburg-Jackson: dt = 0.025
#pragma unroll
    for (int i = 0; i < EPT; ++i) ua[i] = u0[i];
    rk4_2nd(ua, 0.5f * DT0, 3, s_u, tid, HJRhs{});
#pragma unroll
    for (int i = 0; i < EPT; ++i) acc[i] += w2 * ua[i];

    // store evolved as f16, row-major (M = b*T+t, K = d) — ready for WMMA A
#pragma unroll
    for (int i = 0; i < EPT; ++i)
        evh[((size_t)b * T_DIM + tid + i * NTH) * D_DIM + d] = (_Float16)acc[i];
}

// ------------------------- Kernel 3: WMMA f16 GEMM --------------------------
// out[m,n] = sum_k A[m,k]*W[n,k] + bias[n];  M=16384, N=K=1024.
// CDNA5 16-bit operand layout: lane L holds row (L&15); K-halves
// {0..7,16..23} (L<16) or {8..15,24..31} (L>=16) -> two b128 loads per frag.

__device__ __forceinline__ v16h load_frag(const _Float16* p) {
    v8h lo = *(const v8h*)p;          // K-chunk +0..7
    v8h hh = *(const v8h*)(p + 16);   // K-chunk +16..23
    return __builtin_shufflevector(lo, hh, 0, 1, 2, 3, 4, 5, 6, 7,
                                           8, 9, 10, 11, 12, 13, 14, 15);
}

__global__ __launch_bounds__(256) void gemm_kernel(const _Float16* __restrict__ A,
                                                   const _Float16* __restrict__ Bw,
                                                   const float* __restrict__ bias,
                                                   float* __restrict__ out) {
    const int lane = threadIdx.x & 31;
    const int wave = threadIdx.x >> 5;
    const int r    = lane & 15;
    const int hi   = lane >> 4;
    const int m0   = blockIdx.x * 256 + wave * 32;   // each wave: 32(M) x 64(N)
    const int n0   = blockIdx.y * 64;
    const int K = D_DIM, N = D_DIM;

    const _Float16* ap[2];
    const _Float16* bp[4];
#pragma unroll
    for (int mi = 0; mi < 2; ++mi) ap[mi] = A  + (size_t)(m0 + mi * 16 + r) * K + 8 * hi;
#pragma unroll
    for (int nj = 0; nj < 4; ++nj) bp[nj] = Bw + (size_t)(n0 + nj * 16 + r) * K + 8 * hi;

    v8f c[2][4];
#pragma unroll
    for (int mi = 0; mi < 2; ++mi)
#pragma unroll
        for (int nj = 0; nj < 4; ++nj) c[mi][nj] = (v8f){};

    for (int k0 = 0; k0 < K; k0 += 32) {
        v16h af[2], bf[4];
#pragma unroll
        for (int mi = 0; mi < 2; ++mi) af[mi] = load_frag(ap[mi] + k0);
#pragma unroll
        for (int nj = 0; nj < 4; ++nj) bf[nj] = load_frag(bp[nj] + k0);
#pragma unroll
        for (int mi = 0; mi < 2; ++mi)
#pragma unroll
            for (int nj = 0; nj < 4; ++nj)
                c[mi][nj] = __builtin_amdgcn_wmma_f32_16x16x32_f16(
                    false, af[mi], false, bf[nj], (short)0, c[mi][nj], false, false);
    }

    // C/D layout: VGPR r2 -> M = r2 + 8*hi, N = lane&15
#pragma unroll
    for (int mi = 0; mi < 2; ++mi) {
#pragma unroll
        for (int nj = 0; nj < 4; ++nj) {
            const int n = n0 + nj * 16 + r;
            const float bv = bias[n];
#pragma unroll
            for (int r2 = 0; r2 < 8; ++r2) {
                const int m = m0 + mi * 16 + 8 * hi + r2;
                out[(size_t)m * N + n] = c[mi][nj][r2] + bv;
            }
        }
    }
}

// ------------------------- Kernel 1: W -> f16 -------------------------------
__global__ void wconv_kernel(const float* __restrict__ W, _Float16* __restrict__ Wh, int n) {
    int i = blockIdx.x * blockDim.x + threadIdx.x;
    if (i < n) Wh[i] = (_Float16)W[i];
}

// ---------------------------------------------------------------------------
extern "C" void kernel_launch(void* const* d_in, const int* in_sizes, int n_in,
                              void* d_out, int out_size, void* d_ws, size_t ws_size,
                              hipStream_t stream) {
    const float* x    = (const float*)d_in[0];
    const float* mix  = (const float*)d_in[1];
    const float* W    = (const float*)d_in[2];
    const float* bias = (const float*)d_in[3];
    float* out = (float*)d_out;

    // workspace: evolved f16 (32 MB) | W f16 (2 MB)
    _Float16* evh = (_Float16*)d_ws;
    _Float16* Wh  = (_Float16*)((char*)d_ws +
                     (size_t)B_DIM * T_DIM * D_DIM * sizeof(_Float16));

    wconv_kernel<<<(D_DIM * D_DIM + 255) / 256, 256, 0, stream>>>(W, Wh, D_DIM * D_DIM);
    pde_kernel<<<B_DIM * D_DIM, NTH, 0, stream>>>(x, mix, evh);
    dim3 g((B_DIM * T_DIM) / 256, D_DIM / 64);
    gemm_kernel<<<g, 256, 0, stream>>>(evh, Wh, bias, out);
}
